// AdaTTSp_15058155340400
// MI455X (gfx1250) — compile-verified, matrix operand order checked by reading, not run on hardware
//
#include <hip/hip_runtime.h>

#define DEV __device__ __forceinline__

typedef __attribute__((ext_vector_type(4)))  __bf16 v4bf;
typedef __attribute__((ext_vector_type(16))) __bf16 v16bf;
typedef __attribute__((ext_vector_type(8)))  float  v8f;

constexpr int L_ = 2, T_ = 8, E_ = 2, D_ = 128, H_ = 128, NE_ = 16;
constexpr int MT = 32;                 // samples per block (2 M-subtiles per wave)
constexpr int TILE_ELEMS = 128 * 128;  // one staged weight tile (bf16 elements)
constexpr int TILE_BYTES = TILE_ELEMS * 2;          // 32KB
constexpr int TILES_PER_LAYER = 1 + 2 * NE_;        // gate + (w1,w2) x 16 experts
constexpr int NTILES = L_ * TILES_PER_LAYER;        // 66

union ABf { v16bf v; v4bf c[4]; };

DEV unsigned short f2bf(float f) {
  unsigned int u = __float_as_uint(f);
  u += 0x7FFFu + ((u >> 16) & 1u);     // round-to-nearest-even
  return (unsigned short)(u >> 16);
}

// CDNA5 16-bit A/B operand tile from LDS row-major [row][128] bf16:
// lanes 0-15: K = kb+0..7, kb+16..23 ; lanes 16-31: +8 (off).
DEV v16bf load_tile(const unsigned short* base, int row, int kb, int off) {
  const unsigned short* p = base + row * 128 + kb + off;
  ABf r;
  r.c[0] = *reinterpret_cast<const v4bf*>(p);
  r.c[1] = *reinterpret_cast<const v4bf*>(p + 4);
  r.c[2] = *reinterpret_cast<const v4bf*>(p + 16);
  r.c[3] = *reinterpret_cast<const v4bf*>(p + 20);
  return r.v;
}

// Two 16x16 C tiles (M-subtiles 0..15 / 16..31) sharing B-tile loads:
// 8 chained bf16 WMMAs over K=128.
DEV void gemm128x2(const unsigned short* aB, int ar0, int ar1,
                   const unsigned short* bB, int br, int off,
                   v8f& acc0, v8f& acc1) {
#pragma unroll
  for (int kb = 0; kb < 128; kb += 32) {
    v16bf b  = load_tile(bB, br, kb, off);
    v16bf a0 = load_tile(aB, ar0, kb, off);
    v16bf a1 = load_tile(aB, ar1, kb, off);
    acc0 = __builtin_amdgcn_wmma_f32_16x16x32_bf16(false, a0, false, b,
                                                   (short)0, acc0, false, false);
    acc1 = __builtin_amdgcn_wmma_f32_16x16x32_bf16(false, a1, false, b,
                                                   (short)0, acc1, false, false);
  }
}

// Async-DMA one 32KB bf16 tile global->LDS; 8 x b128 per thread (ASYNCcnt +8).
DEV void async_tile_issue(const unsigned short* gsrc, unsigned lds_off, int tid) {
  const char* g = (const char*)gsrc + tid * 16;
  unsigned     l = lds_off + tid * 16;
#pragma unroll
  for (int c = 0; c < 8; ++c) {
    asm volatile("global_load_async_to_lds_b128 %0, %1, off"
                 :: "v"(l + c * 4096), "v"(g + c * 4096) : "memory");
  }
}

DEV void async_wait_newer(int newer) {   // wait until only `newer` tiles outstanding
  if (newer >= 1) asm volatile("s_wait_asynccnt 8" ::: "memory");
  else            asm volatile("s_wait_asynccnt 0" ::: "memory");
}

// ---------------- prep: fp32 weights -> pre-transposed bf16 tiles in ws ---------
// ws tile order (per layer l): [gate_wT(l)] [w1T(l,0)] [w2T(l,0)] ... [w2T(l,15)]
// gate tile rows = t*16+n, expert tile rows = n; cols = k (contiguous).
__global__ __launch_bounds__(256)
void adatt_prep_kernel(const float* __restrict__ w1, const float* __restrict__ w2,
                       const float* __restrict__ gate_w, unsigned short* __restrict__ ws) {
  int idx = blockIdx.x * 256 + threadIdx.x;
  int tile = idx >> 14;
  int within = idx & (TILE_ELEMS - 1);
  int row = within >> 7, k = within & 127;
  int l = tile / TILES_PER_LAYER, j = tile % TILES_PER_LAYER;
  float v;
  if (j == 0) {
    int t = row >> 4, n = row & 15;
    v = gate_w[((size_t)(l * T_ + t) * D_ + k) * NE_ + n];
  } else {
    int e = (j - 1) >> 1;
    const float* w = ((j - 1) & 1) ? w2 : w1;
    v = w[((size_t)(l * NE_ + e) * D_ + k) * H_ + row];
  }
  ws[idx] = f2bf(v);
}

// ---------------- main fused kernel --------------------------------------------
__global__ __launch_bounds__(256)
void adatt_fused_kernel(const float* __restrict__ inputs,
                        const float* __restrict__ b1,
                        const float* __restrict__ b2,
                        const float* __restrict__ gate_b,
                        const float* __restrict__ sew,
                        const unsigned short* __restrict__ wtiles,
                        float* __restrict__ out) {
  __shared__ __align__(16) unsigned short lds_x[T_ * MT * 128];   // [t][m][k]  64KB
  __shared__ __align__(16) unsigned short lds_w[2 * TILE_ELEMS];  // 2-deep ring 64KB
  __shared__ __align__(16) unsigned short lds_h[MT * 128];        // [m][k]      8KB
  __shared__ __align__(16) float lds_g[MT * T_ * NE_];            // [m][t][e]  16KB

  const int tid  = threadIdx.x;
  const int lane = tid & 31;
  const int wv   = tid >> 5;
  const int nl   = lane & 15;
  const int hi   = (lane >> 4) & 1;
  const int off  = hi * 8;
  const long b0  = (long)blockIdx.x * MT;

  const unsigned ldsw_base = (unsigned)(size_t)&lds_w[0];  // LDS byte offset (addr[31:0])

  // kick off the weight-tile pipeline (depth 1) before anything else
  async_tile_issue(wtiles, ldsw_base, tid);
  int gi = 0;

  // overlapped with async DMA: stage layer-0 x tile (fp32 -> bf16 in LDS)
  {
    const float* xs = inputs + b0 * T_ * D_;
#pragma unroll
    for (int i = 0; i < 32; ++i) {
      int g  = (i << 8) | tid;              // 0..8191 groups of 4 along k
      int k4 = (g & 31) << 2;
      int m  = (g >> 5) & 31;
      int t  = g >> 10;
      const float4 v = *reinterpret_cast<const float4*>(xs + (m * T_ + t) * D_ + k4);
      unsigned int p0 = (unsigned)f2bf(v.x) | ((unsigned)f2bf(v.y) << 16);
      unsigned int p1 = (unsigned)f2bf(v.z) | ((unsigned)f2bf(v.w) << 16);
      *reinterpret_cast<uint2*>(&lds_x[(t * MT + m) * 128 + k4]) = make_uint2(p0, p1);
    }
  }

  // barrier; prefetch tile gi+1 into the slot vacated by tile gi-1;
  // wait for tile gi; barrier so all waves see it.
  auto tile_sync = [&]() {
    __syncthreads();
    if (gi + 1 < NTILES)
      async_tile_issue(wtiles + (size_t)(gi + 1) * TILE_ELEMS,
                       ldsw_base + (unsigned)((gi + 1) & 1) * TILE_BYTES, tid);
    async_wait_newer(NTILES - 1 - gi);
    __syncthreads();
  };

  float facc[T_][2][8];

  for (int l = 0; l < L_; ++l) {
    // ---- gating: wave wv = task; logits = [32 samples x 16 experts] (2 C tiles)
    tile_sync();
    {
      const unsigned short* gwb = &lds_w[(gi & 1) * TILE_ELEMS];
      const int t = wv;
      v8f acc0 = {}, acc1 = {};
      gemm128x2(&lds_x[t * MT * 128], nl, 16 + nl, gwb, t * 16 + nl, off, acc0, acc1);
      float gb = gate_b[((size_t)l * T_ + t) * NE_ + nl];
#pragma unroll
      for (int s = 0; s < 2; ++s) {
#pragma unroll
        for (int r = 0; r < 8; ++r) {
          float v  = (s ? acc1[r] : acc0[r]) + gb;
          float mx = v;
#pragma unroll
          for (int sh = 1; sh < 16; sh <<= 1) mx = fmaxf(mx, __shfl_xor(mx, sh, 32));
          float ex = __expf(v - mx);
          float sm = ex;
#pragma unroll
          for (int sh = 1; sh < 16; sh <<= 1) sm += __shfl_xor(sm, sh, 32);
          lds_g[((s * 16 + r + hi * 8) * T_ + t) * NE_ + nl] = ex / sm;
        }
      }
    }
    gi++;

#pragma unroll
    for (int t2 = 0; t2 < T_; ++t2)
#pragma unroll
      for (int s = 0; s < 2; ++s)
#pragma unroll
        for (int r = 0; r < 8; ++r) facc[t2][s][r] = 0.f;

    // ---- expert loop
    for (int e = 0; e < NE_; ++e) {
      const int t = e >> 1;
      // h = relu(x_t @ w1 + b1): wave wv owns output cols wv*16..+15
      tile_sync();
      {
        const unsigned short* wb = &lds_w[(gi & 1) * TILE_ELEMS];
        v8f acc0 = {}, acc1 = {};
        gemm128x2(&lds_x[t * MT * 128], nl, 16 + nl, wb, wv * 16 + nl, off, acc0, acc1);
        float bv = b1[(size_t)(l * NE_ + e) * H_ + wv * 16 + nl];
#pragma unroll
        for (int r = 0; r < 8; ++r) {
          lds_h[(r + hi * 8) * 128 + wv * 16 + nl]      = f2bf(fmaxf(acc0[r] + bv, 0.f));
          lds_h[(16 + r + hi * 8) * 128 + wv * 16 + nl] = f2bf(fmaxf(acc1[r] + bv, 0.f));
        }
      }
      gi++;
      // eo = relu(h @ w2 + b2); fuse gated sum + self-expert residual into facc
      tile_sync();
      {
        const unsigned short* wb = &lds_w[(gi & 1) * TILE_ELEMS];
        v8f acc0 = {}, acc1 = {};
        gemm128x2(lds_h, nl, 16 + nl, wb, wv * 16 + nl, off, acc0, acc1);
        float bv = b2[(size_t)(l * NE_ + e) * H_ + wv * 16 + nl];
        float sv = sew[((size_t)l * T_ + t) * E_ + (e & 1)];
        float eo[2][8];
#pragma unroll
        for (int r = 0; r < 8; ++r) {
          eo[0][r] = fmaxf(acc0[r] + bv, 0.f);
          eo[1][r] = fmaxf(acc1[r] + bv, 0.f);
        }
#pragma unroll
        for (int t2 = 0; t2 < T_; ++t2) {
#pragma unroll
          for (int s = 0; s < 2; ++s) {
#pragma unroll
            for (int r = 0; r < 8; ++r) {
              float gt = lds_g[((s * 16 + r + hi * 8) * T_ + t2) * NE_ + e];
              if (t2 == t) gt += sv;
              facc[t2][s][r] = fmaf(gt, eo[s][r], facc[t2][s][r]);
            }
          }
        }
      }
      gi++;
    }
    __syncthreads();   // all lds_x readers done before overwrite

    if (l + 1 < L_) {
#pragma unroll
      for (int t2 = 0; t2 < T_; ++t2)
#pragma unroll
        for (int s = 0; s < 2; ++s)
#pragma unroll
          for (int r = 0; r < 8; ++r)
            lds_x[(t2 * MT + s * 16 + r + hi * 8) * 128 + wv * 16 + nl] =
                f2bf(facc[t2][s][r]);
      // visibility covered by the barriers in the next layer's first tile_sync
    } else {
#pragma unroll
      for (int t2 = 0; t2 < T_; ++t2) {
#pragma unroll
        for (int s = 0; s < 2; ++s) {
#pragma unroll
          for (int r = 0; r < 8; ++r) {
            long row = b0 + s * 16 + r + hi * 8;
            out[(row * T_ + t2) * H_ + wv * 16 + nl] = facc[t2][s][r];
          }
        }
      }
    }
  }
}

extern "C" void kernel_launch(void* const* d_in, const int* in_sizes, int n_in,
                              void* d_out, int out_size, void* d_ws, size_t ws_size,
                              hipStream_t stream) {
  const float* inputs = (const float*)d_in[0];
  const float* w1     = (const float*)d_in[1];
  const float* b1     = (const float*)d_in[2];
  const float* w2     = (const float*)d_in[3];
  const float* b2     = (const float*)d_in[4];
  const float* gate_w = (const float*)d_in[5];
  const float* gate_b = (const float*)d_in[6];
  const float* sew    = (const float*)d_in[7];
  float* out = (float*)d_out;
  unsigned short* ws = (unsigned short*)d_ws;   // needs NTILES*TILE_ELEMS*2 = ~2.1MB

  // 1) one-off: weights -> pre-transposed bf16 tiles in consumption order
  {
    int total = NTILES * TILE_ELEMS;
    adatt_prep_kernel<<<dim3(total / 256), dim3(256), 0, stream>>>(w1, w2, gate_w, ws);
  }
  // 2) fused two-layer AdaTT with async-DMA weight pipeline
  {
    const int Bn = in_sizes[0] / (T_ * D_);     // 32768
    adatt_fused_kernel<<<dim3(Bn / MT), dim3(256), 0, stream>>>(
        inputs, b1, b2, gate_b, sew, ws, out);
  }
}
